// GNN_47313359732963
// MI455X (gfx1250) — compile-verified, matrix operand order checked by reading, not run on hardware
//
#include <hip/hip_runtime.h>

// ---------------------------------------------------------------------------
// GIN-style GNN forward for MI455X (gfx1250, wave32, WMMA).
// Memory-bound (h is 12.8MB -> L2-resident in 192MB L2); GEMMs routed through
// v_wmma_f32_16x16x32_bf16 so matrix work is negligible vs edge traffic.
// Tail handling: boundary test derived purely from blockIdx (SGPR-scalar) so
// interior blocks take an s_cbranch fast path with unconditional saddr stores.
// ---------------------------------------------------------------------------

#define NN      50000
#define EE      800000
#define ET      (EE + NN)      // edges + self loops
#define EMB     64
#define ATOM_F  32
#define BOND_F  16
#define NLAYER  5

typedef __attribute__((ext_vector_type(16))) __bf16 v16bf;
typedef __attribute__((ext_vector_type(8)))  float  v8f;

// --- WMMA helpers (layouts per CDNA5 ISA §7.12.2) ----------------
// 16-bit A (16x32): lane -> row = lane&15, kbase = (lane>>4)*8;
// elements 0..7 -> K = kbase+e ; elements 8..15 -> K = kbase+16+e.
// B (32x16) mirrored with N = lane&15 (weights row-major over K -> contiguous).

__device__ __forceinline__ v16bf load_tile_f32(const float* rowptr, int kbase) {
  v16bf a;
#pragma unroll
  for (int e = 0; e < 8; ++e) a[e] = (__bf16)rowptr[kbase + e];
#pragma unroll
  for (int e = 0; e < 8; ++e) a[8 + e] = (__bf16)rowptr[kbase + 16 + e];
  return a;
}

__device__ __forceinline__ v16bf load_tile_bf16(const __bf16* rowptr, int kbase) {
  v16bf a;
#pragma unroll
  for (int e = 0; e < 8; ++e) a[e] = rowptr[kbase + e];
#pragma unroll
  for (int e = 0; e < 8; ++e) a[8 + e] = rowptr[kbase + 16 + e];
  return a;
}

__device__ __forceinline__ v8f wmma_bf16(v16bf a, v16bf b, v8f c) {
  // (neg_a, A, neg_b, B, c_mod, C, reuse_a, reuse_b)
  return __builtin_amdgcn_wmma_f32_16x16x32_bf16(false, a, false, b,
                                                 (short)0, c, false, false);
}

// --- init: zero aggregation buffer + BN stats -------------------------------
__global__ void zero_kernel(float* __restrict__ aggr, float* __restrict__ stats,
                            int n_aggr) {
  int i = blockIdx.x * blockDim.x + threadIdx.x;
  if (i < n_aggr) aggr[i] = 0.0f;
  if (i < 2 * EMB) stats[i] = 0.0f;
}

// --- h = x @ Wemb^T (one K=32 WMMA chunk per tile) --------------------------
__global__ __launch_bounds__(128) void embed_kernel(
    const float* __restrict__ x, const float* __restrict__ Wemb,
    float* __restrict__ h) {
  const int tid  = threadIdx.x;
  const int wave = tid >> 5;
  const int lane = tid & 31;
  const int half = lane >> 4;
  const int lm   = lane & 15;
  const int row_base = blockIdx.x * 64 + wave * 16;

  // Block-scalar tail test (blockIdx only) -> s_cmp/s_cbranch, no exec churn.
  const bool full = (blockIdx.x * 64 + 64) <= NN;

  int r  = row_base + lm;
  int rc = (r < NN) ? r : (NN - 1);                 // clamp; stores are masked
  v16bf a = load_tile_f32(x + rc * ATOM_F, half * 8);

#pragma unroll
  for (int nt = 0; nt < 4; ++nt) {
    const int n0 = nt * 16;
    v16bf b = load_tile_f32(Wemb + (n0 + lm) * ATOM_F, half * 8);
    v8f c = {};
    c = wmma_bf16(a, b, c);
    const int base = (row_base + half * 8) * EMB + n0 + lm;
    if (full) {
#pragma unroll
      for (int v = 0; v < 8; ++v) h[base + v * EMB] = c[v];
    } else {
#pragma unroll
      for (int v = 0; v < 8; ++v) {
        int m = row_base + half * 8 + v;            // C/D: M = vgpr + 8*half
        if (m < NN) h[m * EMB + n0 + lm] = c[v];
      }
    }
  }
}

// --- fused edge embedding + gather + scatter-add ----------------------------
// 64 threads per edge (one per output feature); We staged conflict-free in LDS.
__global__ __launch_bounds__(256) void edge_scatter_kernel(
    const float* __restrict__ h, const float* __restrict__ edge_attr,
    const int* __restrict__ srcA, const int* __restrict__ dstA,
    const float* __restrict__ We, const float* __restrict__ be,
    float* __restrict__ aggr) {
  __shared__ float lWe[BOND_F * EMB];   // [k*64 + f]
  __shared__ float lbe[EMB];
  const int tid = threadIdx.x;
  for (int j = tid; j < EMB * BOND_F; j += 256) {
    int f = j >> 4, k = j & 15;
    lWe[k * EMB + f] = We[j];
  }
  if (tid < EMB) lbe[tid] = be[tid];
  __syncthreads();

  const int f = tid & 63;
  const int e = blockIdx.x * 4 + (tid >> 6);
  if (e >= ET) return;

  int si, di;
  float ee;
  if (e < EE) {
    si = srcA[e];
    di = dstA[e];
    if (e + 4 < EE)
      __builtin_prefetch(edge_attr + (e + 4) * BOND_F, 0, 0);
    const float4* ea4 = (const float4*)(edge_attr + e * BOND_F);
    float4 q0 = ea4[0], q1 = ea4[1], q2 = ea4[2], q3 = ea4[3];
    float acc = lbe[f];
    acc = fmaf(q0.x, lWe[ 0 * EMB + f], acc);
    acc = fmaf(q0.y, lWe[ 1 * EMB + f], acc);
    acc = fmaf(q0.z, lWe[ 2 * EMB + f], acc);
    acc = fmaf(q0.w, lWe[ 3 * EMB + f], acc);
    acc = fmaf(q1.x, lWe[ 4 * EMB + f], acc);
    acc = fmaf(q1.y, lWe[ 5 * EMB + f], acc);
    acc = fmaf(q1.z, lWe[ 6 * EMB + f], acc);
    acc = fmaf(q1.w, lWe[ 7 * EMB + f], acc);
    acc = fmaf(q2.x, lWe[ 8 * EMB + f], acc);
    acc = fmaf(q2.y, lWe[ 9 * EMB + f], acc);
    acc = fmaf(q2.z, lWe[10 * EMB + f], acc);
    acc = fmaf(q2.w, lWe[11 * EMB + f], acc);
    acc = fmaf(q3.x, lWe[12 * EMB + f], acc);
    acc = fmaf(q3.y, lWe[13 * EMB + f], acc);
    acc = fmaf(q3.z, lWe[14 * EMB + f], acc);
    acc = fmaf(q3.w, lWe[15 * EMB + f], acc);
    ee = acc;
  } else {                                 // self loop: edge_attr == 0
    si = di = e - EE;
    ee = lbe[f];
  }
  float msg = h[si * EMB + f] + ee;
  atomicAdd(&aggr[di * EMB + f], msg);
}

// --- node MLP: relu(aggr@W1^T+b1)@W2^T+b2, + BN partial sums ----------------
// 4 waves, 64 rows per workgroup; intermediate hid kept as bf16 in LDS.
__global__ __launch_bounds__(128) void node_mlp_kernel(
    const float* __restrict__ aggr,
    const float* __restrict__ W1, const float* __restrict__ b1,
    const float* __restrict__ W2, const float* __restrict__ b2,
    float* __restrict__ hn, float* __restrict__ stats) {
  __shared__ __bf16 hidL[64 * 2 * EMB];   // 64 rows x 128 cols, 16KB
  __shared__ float  sSum[EMB], sSq[EMB];

  const int tid  = threadIdx.x;
  const int wave = tid >> 5;
  const int lane = tid & 31;
  const int half = lane >> 4;
  const int lm   = lane & 15;
  const int row_base = blockIdx.x * 64;
  // Block-scalar tail test (blockIdx only) -> s_cmp/s_cbranch.
  const bool full = (blockIdx.x * 64 + 64) <= NN;

  if (tid < EMB) { sSum[tid] = 0.0f; sSq[tid] = 0.0f; }

  int r  = row_base + wave * 16 + lm;
  int rc = (r < NN) ? r : (NN - 1);
  const float* arow = aggr + rc * EMB;
  v16bf a0 = load_tile_f32(arow, half * 8);        // K  0..31
  v16bf a1 = load_tile_f32(arow, 32 + half * 8);   // K 32..63

  // GEMM1: wave w -> rows [16w,16w+16), all 128 output features
#pragma unroll
  for (int nt = 0; nt < 8; ++nt) {
    const int n0 = nt * 16;
    const float* wrow = W1 + (n0 + lm) * EMB;
    v16bf b0  = load_tile_f32(wrow, half * 8);
    v16bf b1v = load_tile_f32(wrow, 32 + half * 8);
    v8f c = {};
    c = wmma_bf16(a0, b0, c);
    c = wmma_bf16(a1, b1v, c);
    const float bias = b1[n0 + lm];
#pragma unroll
    for (int v = 0; v < 8; ++v) {
      float t = fmaxf(c[v] + bias, 0.0f);          // relu
      int lr = wave * 16 + half * 8 + v;           // local row
      hidL[lr * (2 * EMB) + n0 + lm] = (__bf16)t;
    }
  }
  __syncthreads();

  // GEMM2: K=128 in 4 chunks, 4 output tiles of 16 per wave
  const __bf16* hrow = hidL + (wave * 16 + lm) * (2 * EMB);
  v16bf h0 = load_tile_bf16(hrow, 0  + half * 8);
  v16bf h1 = load_tile_bf16(hrow, 32 + half * 8);
  v16bf h2 = load_tile_bf16(hrow, 64 + half * 8);
  v16bf h3 = load_tile_bf16(hrow, 96 + half * 8);

#pragma unroll
  for (int nt = 0; nt < 4; ++nt) {
    const int n0 = nt * 16;
    const float* wrow = W2 + (n0 + lm) * (2 * EMB);
    v16bf bb0 = load_tile_f32(wrow, 0  + half * 8);
    v16bf bb1 = load_tile_f32(wrow, 32 + half * 8);
    v16bf bb2 = load_tile_f32(wrow, 64 + half * 8);
    v16bf bb3 = load_tile_f32(wrow, 96 + half * 8);
    v8f c = {};
    c = wmma_bf16(h0, bb0, c);
    c = wmma_bf16(h1, bb1, c);
    c = wmma_bf16(h2, bb2, c);
    c = wmma_bf16(h3, bb3, c);
    const float bias = b2[n0 + lm];
    float s = 0.0f, ss = 0.0f;
    const int base = (row_base + wave * 16 + half * 8) * EMB + n0 + lm;
    if (full) {
#pragma unroll
      for (int v = 0; v < 8; ++v) {
        float t = c[v] + bias;
        hn[base + v * EMB] = t;
        s += t;
        ss += t * t;
      }
    } else {
#pragma unroll
      for (int v = 0; v < 8; ++v) {
        int m = row_base + wave * 16 + half * 8 + v;
        if (m < NN) {
          float t = c[v] + bias;
          hn[m * EMB + n0 + lm] = t;
          s += t;
          ss += t * t;
        }
      }
    }
    atomicAdd(&sSum[n0 + lm], s);
    atomicAdd(&sSq[n0 + lm], ss);
  }
  __syncthreads();
  if (tid < EMB) {
    atomicAdd(&stats[tid], sSum[tid]);
    atomicAdd(&stats[EMB + tid], sSq[tid]);
  }
}

// --- BatchNorm (batch stats, biased var) + optional mish --------------------
__global__ void bn_mish_kernel(const float* __restrict__ hn,
                               const float* __restrict__ stats,
                               const float* __restrict__ gamma,
                               const float* __restrict__ beta,
                               float* __restrict__ out, int apply_mish) {
  int idx = blockIdx.x * blockDim.x + threadIdx.x;
  if (idx >= NN * EMB) return;
  const int col = idx & (EMB - 1);
  const float invN = 1.0f / (float)NN;
  const float mu  = stats[col] * invN;
  const float var = stats[EMB + col] * invN - mu * mu;
  const float rs  = rsqrtf(var + 1e-5f);
  float v = (hn[idx] - mu) * rs * gamma[col] + beta[col];
  if (apply_mish) {
    float sp = (v > 20.0f) ? v : log1pf(__expf(v));  // softplus
    v = v * tanhf(sp);
  }
  out[idx] = v;
}

// ---------------------------------------------------------------------------
extern "C" void kernel_launch(void* const* d_in, const int* in_sizes, int n_in,
                              void* d_out, int out_size, void* d_ws,
                              size_t ws_size, hipStream_t stream) {
  (void)in_sizes; (void)n_in; (void)out_size; (void)ws_size;

  const float* x         = (const float*)d_in[0];
  const float* edge_attr = (const float*)d_in[1];
  const float* Wemb      = (const float*)d_in[2];
  const float* W1        = (const float*)d_in[3];
  const float* b1        = (const float*)d_in[4];
  const float* W2        = (const float*)d_in[5];
  const float* b2        = (const float*)d_in[6];
  const float* We        = (const float*)d_in[7];
  const float* be        = (const float*)d_in[8];
  const float* gamma     = (const float*)d_in[9];
  const float* beta      = (const float*)d_in[10];
  const int*   ei        = (const int*)d_in[11];
  const int*   srcA      = ei;           // edge_index[0]
  const int*   dstA      = ei + EE;      // edge_index[1]

  // workspace: h ping (N*64) | aggr (N*64) | stats (128)  ~= 25.6 MB
  float* ws    = (float*)d_ws;
  float* hbuf  = ws;
  float* aggr  = ws + (size_t)NN * EMB;
  float* stats = ws + (size_t)2 * NN * EMB;

  const int node_blocks = (NN + 63) / 64;
  const int ew_blocks   = (ET + 3) / 4;
  const int el_blocks   = (NN * EMB + 255) / 256;

  embed_kernel<<<node_blocks, 128, 0, stream>>>(x, Wemb, hbuf);

  for (int l = 0; l < NLAYER; ++l) {
    zero_kernel<<<el_blocks, 256, 0, stream>>>(aggr, stats, NN * EMB);
    edge_scatter_kernel<<<ew_blocks, 256, 0, stream>>>(
        hbuf, edge_attr, srcA, dstA,
        We + (size_t)l * EMB * BOND_F, be + (size_t)l * EMB, aggr);
    // MLP writes pre-BN activations into hbuf (h is dead after edge kernel)
    node_mlp_kernel<<<node_blocks, 128, 0, stream>>>(
        aggr, W1 + (size_t)l * 2 * EMB * EMB, b1 + (size_t)l * 2 * EMB,
        W2 + (size_t)l * EMB * 2 * EMB, b2 + (size_t)l * EMB, hbuf, stats);
    // BN (+mish except last layer); elementwise 1:1 so in-place is safe
    float* outp = (l == NLAYER - 1) ? (float*)d_out : hbuf;
    bn_mish_kernel<<<el_blocks, 256, 0, stream>>>(
        hbuf, stats, gamma + (size_t)l * EMB, beta + (size_t)l * EMB,
        outp, (l < NLAYER - 1) ? 1 : 0);
  }
}
